// vector_quantizer_9165460209941
// MI455X (gfx1250) — matrix operand order repaired and checked by reading, not compiled, not run on hardware
//
#include <hip/hip_runtime.h>
#include <hip/hip_bf16.h>

// Problem constants (match reference)
#define Bdim 32
#define Ddim 64
#define Ldim 8192
#define Kdim 512
#define Ntot (Bdim * Ldim)        // 262144 rows

// Pre-swizzled B-fragment layout (f16), per 16-code tile nt:
//   half_off = nt*1536 + part*768 + hi*384 + (code&15)*24 + h   (h = 0..15, 8 pad halves)
// part = d>>5 (selects b0/b1), hi = (d>>4)&1, h = d&15.
// Lane chunk stride 24 halves = 48B (16B aligned) -> ds_load_b128 phases land on
// 16 distinct bank groups (2-way conflict floor from 16B alignment).
#define TILE_H 1536
#define PART_H 768
#define HI_H   384
#define MI_H   24
#define BFRAG_HALVES (32 * TILE_H)            // 49152 halves = 98304 bytes
#define BFRAG_BYTES  (BFRAG_HALVES * 2)
#define SMEM_BYTES   (BFRAG_BYTES + Kdim * 4) // + wnorm

typedef __attribute__((ext_vector_type(16))) _Float16     v16h;
typedef __attribute__((ext_vector_type(8)))  float        v8f;
typedef __attribute__((ext_vector_type(4)))  unsigned int v4u;
typedef __attribute__((ext_vector_type(8)))  int          v8i;
typedef __attribute__((ext_vector_type(4)))  int          v4i;

// ---------------------------------------------------------------------------
// Kernel 0: zero scratch accumulators (deterministic per launch)
// ---------------------------------------------------------------------------
__global__ __launch_bounds__(256) void vq_init_kernel(float* __restrict__ loss_acc,
                                                      int* __restrict__ used) {
  if (threadIdx.x == 0) *loss_acc = 0.0f;
  for (int i = threadIdx.x; i < Kdim; i += 256) used[i] = 0;
}

// ---------------------------------------------------------------------------
// Kernel P: one-time codebook prep — f32 -> f16, pre-swizzled into the exact
// per-lane WMMA B-fragment layout (so the hot loop does zero conversions).
// 16384 threads, each converts one pair of consecutive d values -> one u32.
// ---------------------------------------------------------------------------
__global__ __launch_bounds__(256) void vq_prep_kernel(const float* __restrict__ w,
                                                      unsigned int* __restrict__ bfrag_u32) {
  const int t    = blockIdx.x * 256 + threadIdx.x;   // 0..16383
  const int code = t >> 5;                           // 512 codes
  const int d0   = (t & 31) * 2;                     // even d
  const float w0 = w[code * Ddim + d0];
  const float w1 = w[code * Ddim + d0 + 1];
  union { _Float16 h[2]; unsigned int u; } pk;
  pk.h[0] = (_Float16)w0;
  pk.h[1] = (_Float16)w1;
  const int nt   = code >> 4;
  const int mi   = code & 15;
  const int part = d0 >> 5;
  const int hi   = (d0 >> 4) & 1;
  const int h    = d0 & 15;
  const int u32_idx = nt * (TILE_H / 2) + part * (PART_H / 2) + hi * (HI_H / 2) + mi * (MI_H / 2) + (h >> 1);
  bfrag_u32[u32_idx] = pk.u;
}

// ---------------------------------------------------------------------------
// Kernel N: one-time per-code squared norms
// ---------------------------------------------------------------------------
__global__ __launch_bounds__(256) void vq_norm_kernel(const float* __restrict__ w,
                                                      float* __restrict__ wnorm) {
  const int k = blockIdx.x * 256 + threadIdx.x;      // 2 blocks -> 512 codes
  float s = 0.0f;
#pragma unroll 16
  for (int d = 0; d < Ddim; ++d) { const float t = w[k * Ddim + d]; s = fmaf(t, t, s); }
  wnorm[k] = s;
}

// ---------------------------------------------------------------------------
// Helper: 1-D TDM DMA (dwords) global -> LDS
// ---------------------------------------------------------------------------
#if __has_builtin(__builtin_amdgcn_tensor_load_to_lds)
__device__ __forceinline__ void tdm_load_1d(unsigned int lds_byte_off,
                                            const void* gsrc,
                                            unsigned int ndwords) {
  const unsigned long long ga = (unsigned long long)(size_t)gsrc;
  v4u g0;
  g0[0] = 1u;                                        // count=1, user descriptor
  g0[1] = lds_byte_off;                              // lds_addr
  g0[2] = (unsigned int)ga;                          // global_addr[31:0]
  g0[3] = (unsigned int)(ga >> 32) | (2u << 30);     // global_addr[56:32] | type=2
  v8i g1;
  g1[0] = (int)(2u << 16);                           // data_size = 4B, no pad, no multicast
  g1[1] = (int)((ndwords & 0xFFFFu) << 16);          // tensor_dim0 lo @ bits 63:48
  g1[2] = (int)((ndwords >> 16) | (1u << 16));       // tensor_dim0 hi | tensor_dim1 = 1
  g1[3] = (int)((unsigned)ndwords << 16);            // tensor_dim1 hi=0 | tile_dim0
  g1[4] = (int)(1);                                  // tile_dim1 = 1 | tile_dim2 = 0
  g1[5] = (int)(ndwords);                            // tensor_dim0_stride
  g1[6] = 0;
  g1[7] = 0;
  v4i gz4; gz4[0] = 0; gz4[1] = 0; gz4[2] = 0; gz4[3] = 0;
  v8i gz8; gz8[0] = 0; gz8[1] = 0; gz8[2] = 0; gz8[3] = 0;
  gz8[4] = 0; gz8[5] = 0; gz8[6] = 0; gz8[7] = 0;
  __builtin_amdgcn_tensor_load_to_lds(g0, g1, gz4, gz4, gz8, 0);
}
#endif

// ---------------------------------------------------------------------------
// Kernel 1: argmin over codes via f16 WMMA.
//   score[n,k] = ||w_k||^2 - 2 * x_n . w_k   (||x_n||^2 constant per row: skipped)
// Block = 256 threads = 8 waves; each wave owns 16 rows, sweeps all 512 codes
// in 16-wide tiles; K-dim 64 done as 2 x WMMA_F32_16X16X32_F16.
// Pre-swizzled f16 fragments + norms are DMA'd into LDS by the TDM (2 descriptors).
// ---------------------------------------------------------------------------
__global__ __launch_bounds__(256) void vq_argmin_kernel(const float* __restrict__ x,
                                                        const unsigned short* __restrict__ bfrag_g,
                                                        const float* __restrict__ wnorm_g,
                                                        int* __restrict__ idx_i,
                                                        float* __restrict__ idx_f) {
  extern __shared__ char smem[];
  _Float16* bfragh = (_Float16*)smem;                     // 98304 B
  float*    wnorm  = (float*)(smem + BFRAG_BYTES);        // 2048 B

  const int tid  = threadIdx.x;
  const int lane = tid & 31;
  const int wave = tid >> 5;                 // 0..7
  const int hi   = lane >> 4;                // half-wave selector
  const int mi   = lane & 15;

#if __has_builtin(__builtin_amdgcn_tensor_load_to_lds)
  if (wave == 0) {
    const unsigned int lds0 = (unsigned int)(size_t)bfragh;
    tdm_load_1d(lds0, bfrag_g, BFRAG_BYTES / 4);          // fragments
    tdm_load_1d(lds0 + BFRAG_BYTES, wnorm_g, Kdim);       // norms
    __builtin_amdgcn_s_wait_tensorcnt(0);
  }
#else
  for (int i = tid; i < (int)(BFRAG_BYTES / 4); i += 256)
    ((unsigned int*)bfragh)[i] = ((const unsigned int*)bfrag_g)[i];
  for (int i = tid; i < Kdim; i += 256) wnorm[i] = wnorm_g[i];
#endif
  __syncthreads();

  // ---- build the two A fragments (16x32 f16 each) for this wave's 16 rows ----
  const int mbase = blockIdx.x * 128 + wave * 16;
  const int nrow  = mbase + mi;
  const int bb    = nrow >> 13;              // / Ldim
  const int ll    = nrow & (Ldim - 1);
  const float* xrow = x + ((size_t)bb * Ddim) * Ldim + ll;

  v16h a0, a1;
#pragma unroll
  for (int h = 0; h < 16; ++h) {
    const int v = h >> 1, p = h & 1;
    const int d0 = hi * 16 + v * 2 + p; // contiguous k order (matches B fragments)
    a0[h] = (_Float16)xrow[(size_t)d0 * Ldim];
    a1[h] = (_Float16)xrow[(size_t)(d0 + 32) * Ldim];
  }

  float best[8];
  int   bidx[8];
#pragma unroll
  for (int r = 0; r < 8; ++r) { best[r] = 3.4e38f; bidx[r] = 0; }

  const int lane_off = hi * HI_H + mi * MI_H;  // this lane's chunk inside a tile

  // ---- sweep 32 code tiles of 16: 2 LDS vector loads + 2 WMMAs per tile ----
#pragma unroll 4
  for (int nt = 0; nt < Kdim / 16; ++nt) {
    const int code = nt * 16 + mi;             // this lane's B column
    const _Float16* bp = bfragh + nt * TILE_H + lane_off;
    const v16h b0 = *(const v16h*)(bp);            // d = 16*hi + 0..15
    const v16h b1 = *(const v16h*)(bp + PART_H);   // d = 32 + 16*hi + 0..15
    v8f acc = {};
    acc = __builtin_amdgcn_wmma_f32_16x16x32_f16(false, a0, false, b0, (short)0, acc, false, false);
    acc = __builtin_amdgcn_wmma_f32_16x16x32_f16(false, a1, false, b1, (short)0, acc, false, false);
    const float wn = wnorm[code];
#pragma unroll
    for (int r = 0; r < 8; ++r) {
      const float sc = fmaf(-2.0f, acc[r], wn);
      if (sc < best[r]) { best[r] = sc; bidx[r] = code; }
    }
  }

  // ---- reduce across the 16 lanes holding each output row; lowest index wins ties ----
#pragma unroll
  for (int r = 0; r < 8; ++r) {
    float s  = best[r];
    int   bi = bidx[r];
#pragma unroll
    for (int off = 8; off >= 1; off >>= 1) {
      const float os = __shfl_xor(s, off, 32);
      const int   oi = __shfl_xor(bi, off, 32);
      if (os < s || (os == s && oi < bi)) { s = os; bi = oi; }
    }
    if (mi == 0) {
      const int row = mbase + r + 8 * hi;    // C/D layout: VGPR r holds rows r and r+8
      idx_i[row] = bi;
      idx_f[row] = (float)bi;
    }
  }
}

// ---------------------------------------------------------------------------
// Kernel 2: gather quantized vectors -> out (straight-through == quantized),
// accumulate sum((w[idx]-x)^2), mark used codes.
// ---------------------------------------------------------------------------
__global__ __launch_bounds__(256) void vq_gather_kernel(const float* __restrict__ x,
                                                        const float* __restrict__ w,
                                                        const int* __restrict__ idx,
                                                        float* __restrict__ out_q,
                                                        float* __restrict__ loss_acc,
                                                        int* __restrict__ used) {
  const int n  = blockIdx.x * 256 + threadIdx.x;     // exact grid: Ntot threads
  const int bb = n >> 13;
  const int ll = n & (Ldim - 1);
  const int k  = idx[n];
  const float4* wr4  = (const float4*)(w + (size_t)k * Ddim);
  const float*  xr   = x + ((size_t)bb * Ddim) * Ldim + ll;
  float*        outr = out_q + ((size_t)bb * Ddim) * Ldim + ll;

  float acc = 0.0f;
#pragma unroll
  for (int q = 0; q < 16; ++q) {
    const float4 wv = wr4[q];
    const float wa[4] = {wv.x, wv.y, wv.z, wv.w};
#pragma unroll
    for (int j = 0; j < 4; ++j) {
      const int d = 4 * q + j;
      const float xv = xr[(size_t)d * Ldim];
      outr[(size_t)d * Ldim] = wa[j];
      const float df = wa[j] - xv;
      acc = fmaf(df, df, acc);
    }
  }
  used[k] = 1;

  __shared__ float red[256];
  red[threadIdx.x] = acc;
  __syncthreads();
  for (int s = 128; s > 0; s >>= 1) {
    if (threadIdx.x < s) red[threadIdx.x] += red[threadIdx.x + s];
    __syncthreads();
  }
  if (threadIdx.x == 0) atomicAdd(loss_acc, red[0]);
}

// ---------------------------------------------------------------------------
// Kernel 3: passthrough copy of the codebook into the output tuple
// ---------------------------------------------------------------------------
__global__ __launch_bounds__(256) void vq_copyw_kernel(const float* __restrict__ w,
                                                       float* __restrict__ out_w) {
  const int i = blockIdx.x * 256 + threadIdx.x;      // 128 blocks * 256 = 32768
  out_w[i] = w[i];
}

// ---------------------------------------------------------------------------
// Kernel 4: finalize loss (= 1.1 * mse) and perplexity (# distinct codes)
// ---------------------------------------------------------------------------
__global__ __launch_bounds__(256) void vq_final_kernel(const float* __restrict__ loss_acc,
                                                       const int* __restrict__ used,
                                                       float* __restrict__ out_loss,
                                                       float* __restrict__ out_perp) {
  __shared__ int cnt[256];
  int c = 0;
  for (int k = threadIdx.x; k < Kdim; k += 256) c += (used[k] != 0);
  cnt[threadIdx.x] = c;
  __syncthreads();
  for (int s = 128; s > 0; s >>= 1) {
    if (threadIdx.x < s) cnt[threadIdx.x] += cnt[threadIdx.x + s];
    __syncthreads();
  }
  if (threadIdx.x == 0) {
    *out_perp = (float)cnt[0];
    *out_loss = 1.1f * (*loss_acc) / (float)((size_t)Ntot * Ddim);
  }
}

// ---------------------------------------------------------------------------
// Host entry
// ---------------------------------------------------------------------------
extern "C" void kernel_launch(void* const* d_in, const int* in_sizes, int n_in,
                              void* d_out, int out_size, void* d_ws, size_t ws_size,
                              hipStream_t stream) {
  const float* x = (const float*)d_in[0];   // [B, D, L]
  const float* w = (const float*)d_in[1];   // [K, D]
  float* out = (float*)d_out;

  // Output tuple layout (flat, return order)
  float* out_q    = out;                                    // B*D*L
  float* out_loss = out + (size_t)Bdim * Ddim * Ldim;       // 1
  float* out_perp = out_loss + 1;                           // 1
  float* out_w    = out_perp + 1;                           // K*D
  float* out_idx  = out_w + (size_t)Kdim * Ddim;            // N

  // Scratch layout
  char* ws = (char*)d_ws;
  int*          ws_idx   = (int*)ws;                                   // N ints (1 MiB)
  float*        ws_loss  = (float*)(ws + (size_t)Ntot * 4);            // 1 float (+pad)
  int*          ws_used  = (int*)(ws + (size_t)Ntot * 4 + 16);         // 512 ints
  size_t        off_bf   = ((size_t)Ntot * 4 + 16 + Kdim * 4 + 255) & ~(size_t)255;
  unsigned int* ws_bfrag = (unsigned int*)(ws + off_bf);               // 98304 B pre-swizzled f16
  float*        ws_wnorm = (float*)(ws + off_bf + BFRAG_BYTES);        // 2048 B

  hipLaunchKernelGGL(vq_init_kernel, dim3(1), dim3(256), 0, stream, ws_loss, ws_used);
  hipLaunchKernelGGL(vq_prep_kernel, dim3(64), dim3(256), 0, stream, w, ws_bfrag);
  hipLaunchKernelGGL(vq_norm_kernel, dim3(Kdim / 256), dim3(256), 0, stream, w, ws_wnorm);

  (void)hipFuncSetAttribute(reinterpret_cast<const void*>(vq_argmin_kernel),
                            hipFuncAttributeMaxDynamicSharedMemorySize, (int)SMEM_BYTES);
  hipLaunchKernelGGL(vq_argmin_kernel, dim3(Ntot / 128), dim3(256), SMEM_BYTES, stream,
                     x, (const unsigned short*)ws_bfrag, ws_wnorm, ws_idx, out_idx);

  hipLaunchKernelGGL(vq_gather_kernel, dim3(Ntot / 256), dim3(256), 0, stream,
                     x, w, ws_idx, out_q, ws_loss, ws_used);
  hipLaunchKernelGGL(vq_copyw_kernel, dim3((Kdim * Ddim) / 256), dim3(256), 0, stream,
                     w, out_w);
  hipLaunchKernelGGL(vq_final_kernel, dim3(1), dim3(256), 0, stream,
                     ws_loss, ws_used, out_loss, out_perp);
}
